// LFADS_Net_38122129719534
// MI455X (gfx1250) — matrix operand order repaired and checked by reading, not compiled, not run on hardware
//
#include <hip/hip_runtime.h>
#include <stdint.h>
#include <stddef.h>

// ---------------------------------------------------------------------------
// LFADS forward on gfx1250 (MI455X). All matmuls on v_wmma_f32_16x16x32_bf16
// (fp32 accumulate). Every GEMM operand is zero-padded in workspace (K to
// mult-of-32, weight rows to mult-of-16), so the WMMA K-loop is branch-free:
// 6x global_load_b128 + 2x v_wmma per iteration per wave (16x32 tile).
// Dead subgraphs (dyn GRU, f_prior_* heads) are eliminated — they never feed
// `rates`. Workload is sequential-latency bound (~125 GFLOP total, state fits
// in L2), so the design minimizes per-step work and keeps weights bf16-resident.
// Requires ws_size >= ~100 MB.
// ---------------------------------------------------------------------------

constexpr int Bz = 512, Tz = 200, Nz = 200;
constexpr int Ez = 100, Gz = 100, Fz = 20, Uz = 1, Cz = 100;
constexpr int H3 = 300;                 // 3 * hidden(100) for every GRU
constexpr int CAFW = 2 * Ez + Fz;       // 220 controller input width
constexpr float LOGVAR_MIN = -9.210340371976184f; // log(1e-4)

// padded dims
constexpr int KP_N   = 224;  // K=200 or 220 -> 224
constexpr int KP_H   = 128;  // K=100 -> 128
constexpr int KP_U   = 32;   // K=1   -> 32
constexpr int KP_F   = 32;   // K=20  -> 32
constexpr int NP_300 = 304, NP_200 = 208, NP_100 = 112, NP_20 = 32, NP_1 = 16;

typedef __attribute__((ext_vector_type(16))) __bf16 v16bf;
typedef __attribute__((ext_vector_type(8)))  float  v8f;

__device__ __forceinline__ unsigned short f2b(float f) {
  union { float f; uint32_t u; } v; v.f = f;
  uint32_t u = v.u;
  u += 0x7fffu + ((u >> 16) & 1u);      // round-to-nearest-even
  return (unsigned short)(u >> 16);
}
__device__ __forceinline__ float sigm(float x) { return 1.0f / (1.0f + __expf(-x)); }

// ---- strided f32 -> bf16 conversion ---------------------------------------
__global__ void k_cvt2(const float* __restrict__ s, int lds,
                       unsigned short* __restrict__ d, int ldd, int rows, int cols) {
  int i = blockIdx.x * blockDim.x + threadIdx.x;
  if (i >= rows * cols) return;
  int r = i / cols, c = i % cols;
  d[(size_t)r * ldd + c] = f2b(s[(size_t)r * lds + c]);
}

// ---- WMMA GEMM: Y(MxN) = A(MxKp) @ W(NpxKp)^T + bias ----------------------
// A: bf16, lda mult of 32, rows 16B-aligned, K fully padded with zeros.
// W: bf16, Np x ldw (ldw==Kp), zero-padded rows/cols. M must be mult of 16.
// One wave per 16x32 output tile: 1 A-frag, 2 B-frags, 2 accumulators.
// act: 0 none, 1 relu, 2 exp. Yf/Yb optional (fp32 / bf16-strided outputs).
__global__ __launch_bounds__(256)
void k_gemm(const unsigned short* __restrict__ A, int lda,
            const unsigned short* __restrict__ W, int ldw,
            const float* __restrict__ bias,
            float* __restrict__ Yf, int ldyf,
            unsigned short* __restrict__ Yb, int ldyb,
            int M, int Nn, int Kp, int act) {
  int wave = (blockIdx.x * blockDim.x + threadIdx.x) >> 5;
  int lane = threadIdx.x & 31;
  int tn = (Nn + 31) >> 5;
  int tiles = (M >> 4) * tn;
  if (wave >= tiles) return;              // whole-wave exit: EXEC all-1 for WMMA
  int m0 = (wave / tn) << 4;
  int n0 = (wave % tn) << 5;
  int l15 = lane & 15, lh = lane >> 4;

  // A frag (16x32): lane<16 row m0+lane holds K {0..7,16..23}; lane>=16 -> +8
  const unsigned short* Arow = A + (size_t)(m0 + l15) * lda + (lh << 3);
  // B frags (32x16): lane&15 -> column, lane<16 holds K {0..7,16..23}; lane>=16 -> +8
  const unsigned short* W0 = W + (size_t)(n0 + l15) * ldw + (lh << 3);
  const unsigned short* W1 = W0 + (size_t)16 * ldw;

  v8f acc0 = {}, acc1 = {};
  for (int k0 = 0; k0 < Kp; k0 += 32) {
    if (k0 + 32 < Kp) __builtin_prefetch(W0 + k0 + 32, 0, 1);  // global_prefetch_b8
    union { v16bf v; uint4 q[2]; } fa, fb0, fb1;
    fa.q[0]  = *(const uint4*)(Arow + k0);        // K: k0+off   .. +7
    fa.q[1]  = *(const uint4*)(Arow + k0 + 16);   // K: k0+off+16.. +23
    fb0.q[0] = *(const uint4*)(W0 + k0);
    fb0.q[1] = *(const uint4*)(W0 + k0 + 16);
    fb1.q[0] = *(const uint4*)(W1 + k0);
    fb1.q[1] = *(const uint4*)(W1 + k0 + 16);
    acc0 = __builtin_amdgcn_wmma_f32_16x16x32_bf16(false, fa.v, false, fb0.v,
                                                   (short)0, acc0, false, false);
    acc1 = __builtin_amdgcn_wmma_f32_16x16x32_bf16(false, fa.v, false, fb1.v,
                                                   (short)0, acc1, false, false);
  }

  // C/D: VGPR r -> row m0 + r + 8*(lane>>4); col = n0 + (lane&15) (+16)
  int rb = m0 + (lh << 3);
  int col0 = n0 + l15, col1 = col0 + 16;
  v8f accs[2] = {acc0, acc1};
  int cols[2] = {col0, col1};
#pragma unroll
  for (int s = 0; s < 2; ++s) {
    int col = cols[s];
    if (col >= Nn) continue;
    float bv = bias ? bias[col] : 0.0f;
#pragma unroll
    for (int r = 0; r < 8; ++r) {
      float v = accs[s][r] + bv;
      if (act == 1)      v = fmaxf(v, 0.0f);
      else if (act == 2) v = expf(v);
      int row = rb + r;
      if (Yf) Yf[(size_t)row * ldyf + col] = v;
      if (Yb) Yb[(size_t)row * ldyb + col] = f2b(v);
    }
  }
}

// ---- fused GRU gate combine ----------------------------------------------
// clipmode 0: min(h,5) ; 1: clip(h,0,5). Writes fp32 carry (dense), a strided
// bf16 copy, and an optional second strided bf16 destination (concat buffer).
__global__ void k_gru(const float* __restrict__ gi, const float* __restrict__ gh,
                      const float* __restrict__ hprev,
                      float* __restrict__ hout,
                      unsigned short* __restrict__ hb, int ldb,
                      unsigned short* __restrict__ hb2, int ld2,
                      int Bn, int H, int clipmode) {
  int i = blockIdx.x * blockDim.x + threadIdx.x;
  if (i >= Bn * H) return;
  int b = i / H, j = i % H;
  const float* gib = gi + (size_t)b * 3 * H;
  const float* ghb = gh + (size_t)b * 3 * H;
  float r = sigm(gib[j] + ghb[j]);
  float z = sigm(gib[H + j] + ghb[H + j]);
  float n = tanhf(gib[2 * H + j] + r * ghb[2 * H + j]);
  float h = (1.0f - z) * n + z * hprev[i];
  h = fminf(h, 5.0f);
  if (clipmode) h = fmaxf(h, 0.0f);
  hout[i] = h;
  unsigned short hv = f2b(h);
  hb[(size_t)b * ldb + j] = hv;
  if (hb2) hb2[(size_t)b * ld2 + j] = hv;
}

// ---- reparameterization: v = eps * exp(0.5*lv) + mean ---------------------
__global__ void k_reparam(const float* __restrict__ mean, const float* __restrict__ logvar,
                          const float* __restrict__ eps,
                          float* __restrict__ outf,
                          unsigned short* __restrict__ ob1, int ld1,
                          unsigned short* __restrict__ ob2, int ld2,
                          int Bn, int D, int do_clamp) {
  int i = blockIdx.x * blockDim.x + threadIdx.x;
  if (i >= Bn * D) return;
  int b = i / D, d = i % D;
  float lv = logvar[i];
  if (do_clamp) lv = fmaxf(lv, LOGVAR_MIN);
  float v = eps[i] * __expf(0.5f * lv) + mean[i];
  if (outf) outf[i] = v;
  unsigned short bv = f2b(v);
  if (ob1) ob1[(size_t)b * ld1 + d] = bv;
  if (ob2) ob2[(size_t)b * ld2 + d] = bv;
}

// ---------------------------------------------------------------------------
// Host side
// ---------------------------------------------------------------------------
struct Par {
  const float* x;
  const float *egf[4], *egb[4], *ecf[4], *ecb[4], *ctl[4], *gen[4], *dyn[4];
  const float *g0m[2], *g0lv[2], *um[2], *ulv[2], *fm[2], *fv[2],
              *fpm[2], *fpv[2], *l1[2], *l2[2], *l3[2];
  const float *eg0, *eu, *ef;
};

static void resolve(void* const* d_in, const int* sz, int n_in, Par& P) {
  auto F = [&](int i) { return (const float*)d_in[i]; };
  int i; bool psorted;
  if (sz[0] == Bz * Tz * Nz) {          // top-level insertion order: x first
    P.x = F(0); i = 1;
    psorted = (sz[1] == H3 * CAFW);     // sorted params start with ctrl.Wih (300x220)
    P.eg0 = F(51); P.eu = F(52); P.ef = F(53);
  } else {                              // fully tree-sorted: eps_f,eps_g0,eps_u,params,x
    P.ef = F(0); P.eg0 = F(1); P.eu = F(2); i = 3; psorted = true;
    P.x = F(n_in - 1);
  }
  auto take = [&](const float** a, int n) { for (int k = 0; k < n; ++k) a[k] = F(i++); };
  if (!psorted) {
    take(P.egf,4); take(P.egb,4); take(P.ecf,4); take(P.ecb,4);
    take(P.ctl,4); take(P.gen,4); take(P.dyn,4);
    take(P.g0m,2); take(P.g0lv,2); take(P.um,2); take(P.ulv,2);
    take(P.fm,2); take(P.fv,2); take(P.fpm,2); take(P.fpv,2);
    take(P.l1,2); take(P.l2,2); take(P.l3,2);
  } else {
    take(P.ctl,4); take(P.dyn,4); take(P.ecb,4); take(P.ecf,4);
    take(P.egb,4); take(P.egf,4);
    take(P.fm,2); take(P.fpm,2); take(P.fpv,2); take(P.fv,2);
    take(P.g0lv,2); take(P.g0m,2); take(P.gen,4);
    take(P.l1,2); take(P.l2,2); take(P.l3,2);
    take(P.ulv,2); take(P.um,2);
  }
}

static void gemm(hipStream_t st, const unsigned short* A, int lda,
                 const unsigned short* W, int ldw, const float* bias,
                 float* Yf, int ldyf, unsigned short* Yb, int ldyb,
                 int M, int Nn, int Kp, int act) {
  int tiles = (M / 16) * ((Nn + 31) / 32);
  k_gemm<<<(tiles + 7) / 8, 256, 0, st>>>(A, lda, W, ldw, bias,
                                          Yf, ldyf, Yb, ldyb, M, Nn, Kp, act);
}
static void cvtw(hipStream_t st, const float* s, int K, unsigned short* d, int Kp, int rows) {
  int n = rows * K;
  k_cvt2<<<(n + 255) / 256, 256, 0, st>>>(s, K, d, Kp, rows, K);
}

extern "C" void kernel_launch(void* const* d_in, const int* in_sizes, int n_in,
                              void* d_out, int out_size, void* d_ws, size_t ws_size,
                              hipStream_t stream) {
  (void)out_size; (void)ws_size;
  if (n_in < 54) return;
  Par P; resolve(d_in, in_sizes, n_in, P);
  float* dout = (float*)d_out;

  // ---- workspace carve-up: ALL bf16 first (one bulk zeroing memset) -------
  uint8_t* base = (uint8_t*)d_ws;
  size_t off = 0;
  auto AL = [&](size_t bytes) -> void* {
    void* p = base + off; off = (off + bytes + 255) & ~(size_t)255; return p;
  };
  auto AB = [&](size_t elems) { return (unsigned short*)AL(elems * 2); };
  auto AF = [&](size_t elems) { return (float*)AL(elems * 4); };

  // bf16 weights, zero-padded to (Np x Kp)
  unsigned short *w_egf_ih = AB((size_t)NP_300*KP_N), *w_egf_hh = AB((size_t)NP_300*KP_H);
  unsigned short *w_egb_ih = AB((size_t)NP_300*KP_N), *w_egb_hh = AB((size_t)NP_300*KP_H);
  unsigned short *w_ecf_ih = AB((size_t)NP_300*KP_N), *w_ecf_hh = AB((size_t)NP_300*KP_H);
  unsigned short *w_ecb_ih = AB((size_t)NP_300*KP_N), *w_ecb_hh = AB((size_t)NP_300*KP_H);
  unsigned short *w_ctl_ih = AB((size_t)NP_300*KP_N), *w_ctl_hh = AB((size_t)NP_300*KP_H);
  unsigned short *w_gen_ih = AB((size_t)NP_300*KP_U), *w_gen_hh = AB((size_t)NP_300*KP_H);
  unsigned short *w_g0m = AB((size_t)NP_100*KP_N), *w_g0lv = AB((size_t)NP_100*KP_N);
  unsigned short *w_um  = AB((size_t)NP_1*KP_H),   *w_ulv  = AB((size_t)NP_1*KP_H);
  unsigned short *w_fm  = AB((size_t)NP_20*KP_H),  *w_fv   = AB((size_t)NP_20*KP_H);
  unsigned short *w_l1  = AB((size_t)NP_200*KP_F);
  unsigned short *w_l2  = AB((size_t)NP_200*KP_N), *w_l3 = AB((size_t)NP_200*KP_N);

  // bf16 activations (padded K strides)
  unsigned short* caf  = AB((size_t)(Tz + 1) * Bz * KP_N);  // [ef|eb|f|0..] x 224
  unsigned short* xpad = AB((size_t)Bz * Tz * KP_N);        // x -> bf16, 200->224
  unsigned short *hb_egf = AB((size_t)Bz*KP_H), *hb_ecf = AB((size_t)Bz*KP_H);
  unsigned short *hb_egb = AB((size_t)Bz*KP_H), *hb_ecb = AB((size_t)Bz*KP_H);
  unsigned short *egen_b = AB((size_t)Bz*KP_N);
  unsigned short *g_b = AB((size_t)Bz*KP_H), *c_b = AB((size_t)Bz*KP_H);
  unsigned short *u_b = AB((size_t)Bz*KP_U), *f_b = AB((size_t)Bz*KP_F);
  unsigned short *h1_b = AB((size_t)Bz*KP_N), *h2_b = AB((size_t)Bz*KP_N);
  size_t bf16_end = off;

  // fp32 state (zeroed) then fp32 scratch (fully overwritten before reads)
  size_t st_beg = off;
  float *h_egf = AF((size_t)Bz*Ez), *h_ecf = AF((size_t)Bz*Ez);
  float *h_egb = AF((size_t)Bz*Ez), *h_ecb = AF((size_t)Bz*Ez);
  float *c_f = AF((size_t)Bz*Cz);
  size_t st_end = off;
  float *s300a = AF((size_t)Bz*H3), *s300b = AF((size_t)Bz*H3);
  float *s300c = AF((size_t)Bz*H3), *s300d = AF((size_t)Bz*H3);
  float *g0m_f = AF((size_t)Bz*Gz), *g0lv_f = AF((size_t)Bz*Gz);
  float *g_f = AF((size_t)Bz*Gz);
  float *um_f = AF((size_t)Bz*Uz), *ulv_f = AF((size_t)Bz*Uz);
  float *fm_f = AF((size_t)Bz*Fz), *flv_f = AF((size_t)Bz*Fz);

  // ---- 1. zero padded bf16 region + fp32 state ----------------------------
  hipMemsetAsync(base, 0, bf16_end, stream);
  hipMemsetAsync(base + st_beg, 0, st_end - st_beg, stream);

  // ---- 2. weights -> padded bf16, x -> padded bf16 ------------------------
  cvtw(stream, P.egf[0], Nz, w_egf_ih, KP_N, H3);  cvtw(stream, P.egf[1], Ez, w_egf_hh, KP_H, H3);
  cvtw(stream, P.egb[0], Nz, w_egb_ih, KP_N, H3);  cvtw(stream, P.egb[1], Ez, w_egb_hh, KP_H, H3);
  cvtw(stream, P.ecf[0], Nz, w_ecf_ih, KP_N, H3);  cvtw(stream, P.ecf[1], Ez, w_ecf_hh, KP_H, H3);
  cvtw(stream, P.ecb[0], Nz, w_ecb_ih, KP_N, H3);  cvtw(stream, P.ecb[1], Ez, w_ecb_hh, KP_H, H3);
  cvtw(stream, P.ctl[0], CAFW, w_ctl_ih, KP_N, H3);cvtw(stream, P.ctl[1], Cz, w_ctl_hh, KP_H, H3);
  cvtw(stream, P.gen[0], Uz, w_gen_ih, KP_U, H3);  cvtw(stream, P.gen[1], Gz, w_gen_hh, KP_H, H3);
  cvtw(stream, P.g0m[0], 2*Ez, w_g0m, KP_N, Gz);   cvtw(stream, P.g0lv[0], 2*Ez, w_g0lv, KP_N, Gz);
  cvtw(stream, P.um[0],  Cz, w_um,  KP_H, Uz);     cvtw(stream, P.ulv[0],  Cz, w_ulv,  KP_H, Uz);
  cvtw(stream, P.fm[0],  Gz, w_fm,  KP_H, Fz);     cvtw(stream, P.fv[0],   Gz, w_fv,   KP_H, Fz);
  cvtw(stream, P.l1[0],  Fz, w_l1,  KP_F, 200);    cvtw(stream, P.l2[0],   200, w_l2,  KP_N, 200);
  cvtw(stream, P.l3[0],  200, w_l3, KP_N, Nz);
  cvtw(stream, P.x, Nz, xpad, KP_N, Bz * Tz);      // rows indexed (b*T + t)

  int nBE = Bz * Ez, nBG = Bz * Gz, nBC = Bz * Cz;

  // ---- 3. forward encoders (egen_f, econ_f) -------------------------------
  for (int t = 0; t < Tz; ++t) {
    const unsigned short* xt = xpad + (size_t)t * KP_N;   // row stride Tz*KP_N
    gemm(stream, xt, Tz*KP_N, w_egf_ih, KP_N, P.egf[2], s300a, H3, nullptr,0, Bz, H3, KP_N, 0);
    gemm(stream, xt, Tz*KP_N, w_ecf_ih, KP_N, P.ecf[2], s300b, H3, nullptr,0, Bz, H3, KP_N, 0);
    gemm(stream, hb_egf, KP_H, w_egf_hh, KP_H, P.egf[3], s300c, H3, nullptr,0, Bz, H3, KP_H, 0);
    gemm(stream, hb_ecf, KP_H, w_ecf_hh, KP_H, P.ecf[3], s300d, H3, nullptr,0, Bz, H3, KP_H, 0);
    k_gru<<<(nBE+255)/256,256,0,stream>>>(s300a, s300c, h_egf, h_egf, hb_egf, KP_H,
                                          nullptr, 0, Bz, Ez, 0);
    k_gru<<<(nBE+255)/256,256,0,stream>>>(s300b, s300d, h_ecf, h_ecf, hb_ecf, KP_H,
                                          caf + (size_t)t*Bz*KP_N, KP_N, Bz, Ez, 0);
  }
  // ---- 4. backward encoders (egen_b, econ_b) ------------------------------
  for (int t = Tz - 1; t >= 0; --t) {
    const unsigned short* xt = xpad + (size_t)t * KP_N;
    gemm(stream, xt, Tz*KP_N, w_egb_ih, KP_N, P.egb[2], s300a, H3, nullptr,0, Bz, H3, KP_N, 0);
    gemm(stream, xt, Tz*KP_N, w_ecb_ih, KP_N, P.ecb[2], s300b, H3, nullptr,0, Bz, H3, KP_N, 0);
    gemm(stream, hb_egb, KP_H, w_egb_hh, KP_H, P.egb[3], s300c, H3, nullptr,0, Bz, H3, KP_H, 0);
    gemm(stream, hb_ecb, KP_H, w_ecb_hh, KP_H, P.ecb[3], s300d, H3, nullptr,0, Bz, H3, KP_H, 0);
    k_gru<<<(nBE+255)/256,256,0,stream>>>(s300a, s300c, h_egb, h_egb, hb_egb, KP_H,
                                          nullptr, 0, Bz, Ez, 0);
    k_gru<<<(nBE+255)/256,256,0,stream>>>(s300b, s300d, h_ecb, h_ecb, hb_ecb, KP_H,
                                          caf + (size_t)t*Bz*KP_N + Ez, KP_N, Bz, Ez, 0);
  }

  // ---- 5. initial condition g, initial factors f --------------------------
  k_cvt2<<<(nBE+255)/256,256,0,stream>>>(h_egf, Ez, egen_b,      KP_N, Bz, Ez);
  k_cvt2<<<(nBE+255)/256,256,0,stream>>>(h_egb, Ez, egen_b + Ez, KP_N, Bz, Ez);
  gemm(stream, egen_b, KP_N, w_g0m,  KP_N, P.g0m[1],  g0m_f,  Gz, nullptr,0, Bz, Gz, KP_N, 0);
  gemm(stream, egen_b, KP_N, w_g0lv, KP_N, P.g0lv[1], g0lv_f, Gz, nullptr,0, Bz, Gz, KP_N, 0);
  k_reparam<<<(nBG+255)/256,256,0,stream>>>(g0m_f, g0lv_f, P.eg0, g_f, g_b, KP_H,
                                            nullptr, 0, Bz, Gz, 1);
  gemm(stream, g_b, KP_H, w_fm, KP_H, P.fm[1], nullptr, 0,
       caf + (size_t)0*Bz*KP_N + 2*Ez, KP_N, Bz, Fz, KP_H, 0);  // f0 -> caf slot 0

  // ---- 6. generator loop --------------------------------------------------
  for (int t = 0; t < Tz; ++t) {
    unsigned short* caf_t  = caf + (size_t)t * Bz * KP_N;
    unsigned short* fnew_b = caf + (size_t)(t + 1) * Bz * KP_N + 2*Ez; // f_t slot
    // controller GRU (clip to [0,5])
    gemm(stream, caf_t, KP_N, w_ctl_ih, KP_N, P.ctl[2], s300a, H3, nullptr,0, Bz, H3, KP_N, 0);
    gemm(stream, c_b,   KP_H, w_ctl_hh, KP_H, P.ctl[3], s300b, H3, nullptr,0, Bz, H3, KP_H, 0);
    k_gru<<<(nBC+255)/256,256,0,stream>>>(s300a, s300b, c_f, c_f, c_b, KP_H,
                                          nullptr, 0, Bz, Cz, 1);
    // u sample (no logvar clamp)
    gemm(stream, c_b, KP_H, w_um,  KP_H, P.um[1],  um_f,  Uz, nullptr,0, Bz, Uz, KP_H, 0);
    gemm(stream, c_b, KP_H, w_ulv, KP_H, P.ulv[1], ulv_f, Uz, nullptr,0, Bz, Uz, KP_H, 0);
    k_reparam<<<(Bz*Uz+255)/256,256,0,stream>>>(um_f, ulv_f, P.eu + (size_t)t*Bz*Uz,
                                                nullptr, u_b, KP_U, nullptr, 0, Bz, Uz, 0);
    // generator GRU (min with 5)
    gemm(stream, u_b, KP_U, w_gen_ih, KP_U, P.gen[2], s300c, H3, nullptr,0, Bz, H3, KP_U, 0);
    gemm(stream, g_b, KP_H, w_gen_hh, KP_H, P.gen[3], s300d, H3, nullptr,0, Bz, H3, KP_H, 0);
    k_gru<<<(nBG+255)/256,256,0,stream>>>(s300c, s300d, g_f, g_f, g_b, KP_H,
                                          nullptr, 0, Bz, Gz, 0);
    // factor sample (logvar clamped); bf16 into caf slot t+1 AND padded f_b
    gemm(stream, g_b, KP_H, w_fm, KP_H, P.fm[1], fm_f,  Fz, nullptr,0, Bz, Fz, KP_H, 0);
    gemm(stream, g_b, KP_H, w_fv, KP_H, P.fv[1], flv_f, Fz, nullptr,0, Bz, Fz, KP_H, 0);
    k_reparam<<<(Bz*Fz+255)/256,256,0,stream>>>(fm_f, flv_f, P.ef + (size_t)t*Bz*Fz,
                                                nullptr, fnew_b, KP_N, f_b, KP_F, Bz, Fz, 1);
    // rate decoder: relu(lin) -> relu(lin) -> exp(lin), write d_out[b,t,:]
    gemm(stream, f_b,  KP_F, w_l1, KP_F, P.l1[1], nullptr,0, h1_b, KP_N, Bz, 200, KP_F, 1);
    gemm(stream, h1_b, KP_N, w_l2, KP_N, P.l2[1], nullptr,0, h2_b, KP_N, Bz, 200, KP_N, 1);
    gemm(stream, h2_b, KP_N, w_l3, KP_N, P.l3[1], dout + (size_t)t*Nz, Tz*Nz,
         nullptr, 0, Bz, Nz, KP_N, 2);
  }
  // Note: dyn GRU (hd) and f_prior_* heads never influence `rates` -> DCE'd.
}
